// GumbelVectorQuantizer_58858231825069
// MI455X (gfx1250) — matrix operand order repaired and checked by reading, not compiled, not run on hardware
//
#include <hip/hip_runtime.h>

typedef _Float16 v16h __attribute__((ext_vector_type(16)));
typedef _Float16 v8h  __attribute__((ext_vector_type(8)));
typedef _Float16 v4h  __attribute__((ext_vector_type(4)));
typedef float    v8f  __attribute__((ext_vector_type(8)));
typedef unsigned int v4u __attribute__((ext_vector_type(4)));
typedef int          v8i __attribute__((ext_vector_type(8)));
typedef int          v4i __attribute__((ext_vector_type(4)));

#define Bq 8
#define Tq 4096
#define Dq 1024
#define Vq 1024
#define Mrows (Bq * Tq)          // 32768 rows
#define EPSq 1e-20f
#define INV_TEMP 2.0f            // 1 / 0.5

// ---------- prep: codebook f32 -> f16, both orientations (one-time, L2-resident) ----------
__global__ __launch_bounds__(256) void cvt_codebook_kernel(
    const float* __restrict__ cb,      // [Vq, Dq] f32
    _Float16* __restrict__ cbh,        // [Vq, Dq] f16
    _Float16* __restrict__ cbhT)       // [Dq, Vq] f16 (transposed)
{
    const int i4 = blockIdx.x * 256 + threadIdx.x;   // 1M/4 float4 groups
    const int v  = i4 >> 8;                          // 256 float4 per row
    const int d4 = i4 & 255;
    const float4 f = *(const float4*)(cb + (size_t)v * Dq + d4 * 4);
    v4h hv;
    hv[0] = (_Float16)f.x; hv[1] = (_Float16)f.y;
    hv[2] = (_Float16)f.z; hv[3] = (_Float16)f.w;
    *(v4h*)(cbh + (size_t)v * Dq + d4 * 4) = hv;
    const int d = d4 * 4;
    cbhT[(size_t)(d + 0) * Vq + v] = hv[0];
    cbhT[(size_t)(d + 1) * Vq + v] = hv[1];
    cbhT[(size_t)(d + 2) * Vq + v] = hv[2];
    cbhT[(size_t)(d + 3) * Vq + v] = hv[3];
}

// ---------- fused: TDM-stage X, GEMM1, gumbel-softmax, GEMM2 ----------
__global__ __launch_bounds__(256) void gvq_fused_kernel(
    const float* __restrict__ x,        // [Mrows, Dq] f32
    const _Float16* __restrict__ cbh,   // [Vq, Dq] f16
    const _Float16* __restrict__ cbhT,  // [Dq, Vq] f16
    const float* __restrict__ noise,    // [Mrows, Vq] f32
    float* __restrict__ outq,           // [Mrows, Dq]
    float* __restrict__ outs)           // [Mrows, Vq]
{
    // Single 64KB LDS block at offset 0:
    //  phase 0: [0,64K)  = X tile as f32 (TDM destination)
    //  phase 1: [0,32K)  = X tile (then S tile) as f16;  [32K,33K) = reductions
    __shared__ __align__(16) unsigned char smem[65536];
    _Float16 (*hbuf)[Dq] = reinterpret_cast<_Float16(*)[Dq]>(smem);
    float* redmax = (float*)(smem + 32768);          // [8][16]
    float* redsum = (float*)(smem + 32768 + 512);    // [8][16]

    const int tid  = threadIdx.x;
    const int wave = tid >> 5;
    const int lane = tid & 31;
    const int half = lane >> 4;   // 0: lanes 0-15, 1: lanes 16-31
    const int ln   = lane & 15;
    const int row_base = blockIdx.x * 16;

    // ---- stage X tile [16,1024] f32 into LDS via Tensor Data Mover ----
    if (wave == 0) {
        const unsigned long long ga =
            (unsigned long long)(uintptr_t)(x + (size_t)row_base * Dq);
        v4u g0;
        g0[0] = 1u;                                        // count=1, user D#
        g0[1] = 0u;                                        // lds_addr = 0 (smem base)
        g0[2] = (unsigned)(ga & 0xFFFFFFFFull);            // global_addr[31:0]
        g0[3] = (unsigned)((ga >> 32) & 0x01FFFFFFull)     // global_addr[56:32]
              | (2u << 30);                                // type = 2 (image)
        v8i g1;
        g1[0] = (int)(2u << 16);        // data_size = 2 (4 bytes); wg_mask = 0
        g1[1] = (int)(1024u << 16);     // tensor_dim0[15:0]=1024 at bits 63:48
        g1[2] = (int)0x80000000u;       // dim0 hi=0 | tensor_dim1 lo16 = 32768
        g1[3] = (int)(1024u << 16);     // dim1 hi=0 | tile_dim0 = 1024
        g1[4] = 16;                     // tile_dim1 = 16, tile_dim2 = 0
        g1[5] = 1024;                   // tensor_dim0_stride lo32 = 1024
        g1[6] = 0;                      // stride0 hi | stride1 lo
        g1[7] = 0;
        const v4i z4 = {0, 0, 0, 0};
#if __clang_major__ >= 23
        const v8i z8 = {0, 0, 0, 0, 0, 0, 0, 0};
        __builtin_amdgcn_tensor_load_to_lds(g0, g1, z4, z4, z8, 0);
#else
        __builtin_amdgcn_tensor_load_to_lds(g0, g1, z4, z4, 0);
#endif
        __builtin_amdgcn_s_wait_tensorcnt(0);
    }
    __syncthreads();

    // ---- compact f32 -> f16 in place (regs, barrier, halves) ----
    {
        const float4* xf4 = (const float4*)smem;
        float4 tmp[16];
#pragma unroll
        for (int b2 = 0; b2 < 16; ++b2) tmp[b2] = xf4[tid * 16 + b2];
        __syncthreads();
        v4h* h4 = (v4h*)smem;
#pragma unroll
        for (int b2 = 0; b2 < 16; ++b2) {
            v4h hv;
            hv[0] = (_Float16)tmp[b2].x; hv[1] = (_Float16)tmp[b2].y;
            hv[2] = (_Float16)tmp[b2].z; hv[3] = (_Float16)tmp[b2].w;
            h4[tid * 16 + b2] = hv;
        }
    }
    __syncthreads();

    const v8f vzero = {0.f, 0.f, 0.f, 0.f, 0.f, 0.f, 0.f, 0.f};

    // ================= GEMM1: logits = X * C^T =================
    v8f acc[8];
#pragma unroll
    for (int t = 0; t < 8; ++t) acc[t] = vzero;

    const int vwbase = wave * 128;

    for (int kb = 0; kb < Dq / 32; ++kb) {
        const int k0 = kb * 32 + half * 16;
        const v8h alo = *(const v8h*)&hbuf[ln][k0];
        const v8h ahi = *(const v8h*)&hbuf[ln][k0 + 8];
        v16h a;
#pragma unroll
        for (int i = 0; i < 8; ++i) { a[i] = alo[i]; a[8 + i] = ahi[i]; }

#pragma unroll
        for (int t = 0; t < 8; ++t) {
            // B[k][n] = cbh[v][k]; lane = n + 16*(k>=16): 16 contiguous halves
            const int v = vwbase + t * 16 + ln;
            const _Float16* bp = cbh + (size_t)v * Dq + kb * 32 + half * 16;
            __builtin_prefetch(bp + 32, 0, 3);   // global_prefetch_b8 (next k-chunk)
            const v8h blo = *(const v8h*)bp;
            const v8h bhi = *(const v8h*)(bp + 8);
            v16h b;
#pragma unroll
            for (int i = 0; i < 8; ++i) { b[i] = blo[i]; b[8 + i] = bhi[i]; }
            acc[t] = __builtin_amdgcn_wmma_f32_16x16x32_f16(
                false, a, false, b, (short)0, acc[t], false, false);
        }
    }

    // ---- Gumbel noise + temperature ----
    float pmax[8];
#pragma unroll
    for (int j = 0; j < 8; ++j) pmax[j] = -1e30f;

#pragma unroll
    for (int t = 0; t < 8; ++t) {
        const int v = vwbase + t * 16 + ln;
#pragma unroll
        for (int j = 0; j < 8; ++j) {
            const int r = row_base + j + half * 8;
            const float u = noise[(size_t)r * Vq + v];
            const float g = -__logf(-__logf(u + EPSq) + EPSq);
            const float z = (acc[t][j] + g) * INV_TEMP;
            acc[t][j] = z;
            pmax[j] = fmaxf(pmax[j], z);
        }
    }

    // ---- row max across 16 lanes, then cross-wave ----
#pragma unroll
    for (int j = 0; j < 8; ++j) {
#pragma unroll
        for (int msk = 1; msk < 16; msk <<= 1)
            pmax[j] = fmaxf(pmax[j], __shfl_xor(pmax[j], msk, 32));
    }
    if (ln == 0) {
#pragma unroll
        for (int j = 0; j < 8; ++j) redmax[wave * 16 + half * 8 + j] = pmax[j];
    }
    __syncthreads();

    float rmax[8];
#pragma unroll
    for (int j = 0; j < 8; ++j) {
        float mm = redmax[half * 8 + j];
#pragma unroll
        for (int w = 1; w < 8; ++w) mm = fmaxf(mm, redmax[w * 16 + half * 8 + j]);
        rmax[j] = mm;
    }

    // ---- exp + row sum ----
    float psum[8];
#pragma unroll
    for (int j = 0; j < 8; ++j) psum[j] = 0.f;
#pragma unroll
    for (int t = 0; t < 8; ++t) {
#pragma unroll
        for (int j = 0; j < 8; ++j) {
            const float e = __expf(acc[t][j] - rmax[j]);
            acc[t][j] = e;
            psum[j] += e;
        }
    }
#pragma unroll
    for (int j = 0; j < 8; ++j) {
#pragma unroll
        for (int msk = 1; msk < 16; msk <<= 1)
            psum[j] += __shfl_xor(psum[j], msk, 32);
    }
    if (ln == 0) {
#pragma unroll
        for (int j = 0; j < 8; ++j) redsum[wave * 16 + half * 8 + j] = psum[j];
    }
    __syncthreads();

    float rinv[8];
#pragma unroll
    for (int j = 0; j < 8; ++j) {
        float s = redsum[half * 8 + j];
#pragma unroll
        for (int w = 1; w < 8; ++w) s += redsum[w * 16 + half * 8 + j];
        rinv[j] = 1.0f / s;
    }

    // ---- normalize: soft_one_hot -> HBM, and as f16 A-operand into LDS ----
#pragma unroll
    for (int t = 0; t < 8; ++t) {
        const int v = vwbase + t * 16 + ln;
#pragma unroll
        for (int j = 0; j < 8; ++j) {
            const int m = j + half * 8;
            const float p = acc[t][j] * rinv[j];
            outs[(size_t)(row_base + m) * Vq + v] = p;
            hbuf[m][v] = (_Float16)p;
        }
    }
    __syncthreads();

    // ================= GEMM2: quantized = S * C =================
    v8f acc2[8];
#pragma unroll
    for (int t = 0; t < 8; ++t) acc2[t] = vzero;

    const int dwbase = wave * 128;

    for (int kb = 0; kb < Vq / 32; ++kb) {
        const int k0 = kb * 32 + half * 16;
        const v8h alo = *(const v8h*)&hbuf[ln][k0];
        const v8h ahi = *(const v8h*)&hbuf[ln][k0 + 8];
        v16h a;
#pragma unroll
        for (int i = 0; i < 8; ++i) { a[i] = alo[i]; a[8 + i] = ahi[i]; }

#pragma unroll
        for (int t = 0; t < 8; ++t) {
            // B[k][n] = cb[k][d] = cbhT[d][k]: 16 contiguous halves in cbhT row d
            const int d = dwbase + t * 16 + ln;
            const _Float16* bp = cbhT + (size_t)d * Vq + kb * 32 + half * 16;
            __builtin_prefetch(bp + 32, 0, 3);
            const v8h blo = *(const v8h*)bp;
            const v8h bhi = *(const v8h*)(bp + 8);
            v16h b;
#pragma unroll
            for (int i = 0; i < 8; ++i) { b[i] = blo[i]; b[8 + i] = bhi[i]; }
            acc2[t] = __builtin_amdgcn_wmma_f32_16x16x32_f16(
                false, a, false, b, (short)0, acc2[t], false, false);
        }
    }

    // ---- store quantized ----
#pragma unroll
    for (int t = 0; t < 8; ++t) {
        const int d = dwbase + t * 16 + ln;
#pragma unroll
        for (int j = 0; j < 8; ++j) {
            const int r = row_base + j + half * 8;
            outq[(size_t)r * Dq + d] = acc2[t][j];
        }
    }
}

extern "C" void kernel_launch(void* const* d_in, const int* in_sizes, int n_in,
                              void* d_out, int out_size, void* d_ws, size_t ws_size,
                              hipStream_t stream) {
    (void)in_sizes; (void)n_in; (void)out_size; (void)ws_size;
    const float* x     = (const float*)d_in[0];   // inputs [B,T,D] f32
    const float* cb    = (const float*)d_in[1];   // codebook [V,D] f32
    const float* noise = (const float*)d_in[2];   // gumbel_noise [B,T,V] f32
    float* outq = (float*)d_out;                          // quantized [B,T,D]
    float* outs = outq + (size_t)Mrows * Dq;              // soft_one_hot [B,T,V]

    _Float16* cbh  = (_Float16*)d_ws;                     // [V][D] f16 (2 MB)
    _Float16* cbhT = cbh + (size_t)Vq * Dq;               // [D][V] f16 (2 MB)

    // one-time-per-call codebook conversion (idempotent, deterministic)
    cvt_codebook_kernel<<<dim3((Vq * Dq / 4) / 256), dim3(256), 0, stream>>>(cb, cbh, cbhT);

    gvq_fused_kernel<<<dim3(Mrows / 16), dim3(256), 0, stream>>>(
        x, cbh, cbhT, noise, outq, outs);
}
